// Physics_Attention_Structured_Mesh_2D_12506944766453
// MI455X (gfx1250) — compile-verified, matrix operand order checked by reading, not run on hardware
//
#include <hip/hip_runtime.h>
#include <hip/hip_bf16.h>
#include <cstdint>

// ---------------------------------------------------------------------------
// MI455X (gfx1250) implementation.
// Heavy GEMMs (convs as implicit GEMM, token GEMMs, out-proj) use
// v_wmma_f32_16x16x32_bf16 (wave32, 16x16x32 tiles, f32 accum).
// 2D-tile stagings that need no transpose go through the Tensor Data Mover
// (tensor_load_to_lds + s_wait_tensorcnt). The tiny Erwin ball-transformer
// runs one workgroup per ball-tree entirely in LDS.
// ---------------------------------------------------------------------------

typedef __bf16 bf16;
typedef bf16  v16bf __attribute__((ext_vector_type(16)));
typedef bf16  v8bf  __attribute__((ext_vector_type(8)));
typedef float v8f   __attribute__((ext_vector_type(8)));
typedef unsigned int v4u32 __attribute__((ext_vector_type(4)));
typedef int v8i32 __attribute__((ext_vector_type(8)));
typedef int v4i32 __attribute__((ext_vector_type(4)));

#define LANE (threadIdx.x & 31)
#define WAVE ((int)(threadIdx.x >> 5))

// problem dims
constexpr int BN    = 4;
constexpr int HMi   = 128;
constexpr int WMi   = 128;
constexpr int DIMi  = 256;
constexpr int HEADS = 8;
constexpr int DH    = 64;
constexpr int GS    = 64;
constexpr int INNER = HEADS * DH;   // 512
constexpr int NPIX  = HMi * WMi;    // 16384

// ---------------------------------------------------------------------------
// WMMA helpers: tiles stored in LDS as [rows][32] bf16 row-major.
// A-fragment layout (ISA 7.12.2, 16-bit A 16x32): lane m=L%16,
// kbase=(L>=16)?8:0 -> two contiguous 16-byte LDS reads per lane.
// B fragment loaded from the transposed [N][K] tile with the same pattern.
// ---------------------------------------------------------------------------
__device__ inline v16bf frag_ld(const bf16* tile, int row0) {
  int m  = threadIdx.x & 15;
  int kb = ((threadIdx.x >> 4) & 1) << 3;
  const bf16* p = tile + (row0 + m) * 32 + kb;
  union { v16bf v; v8bf h[2]; } u;
  u.h[0] = *(const v8bf*)p;
  u.h[1] = *(const v8bf*)(p + 16);
  return u.v;
}

__device__ inline void wmma_tile4(const bf16* At, const bf16* Bt, int mrow0, v8f acc[4]) {
  v16bf a = frag_ld(At, mrow0);
#pragma unroll
  for (int nt = 0; nt < 4; nt++) {
    v16bf b = frag_ld(Bt, nt * 16);
    acc[nt] = __builtin_amdgcn_wmma_f32_16x16x32_bf16(false, a, false, b,
                                                      (short)0, acc[nt], false, false);
  }
}

__device__ inline void acc_zero(v8f acc[4]) {
#pragma unroll
  for (int i = 0; i < 4; i++)
#pragma unroll
    for (int j = 0; j < 8; j++) acc[i][j] = 0.0f;
}

// ---------------------------------------------------------------------------
// Tensor Data Mover: 2D tile (bf16, data_size=1 => 2-byte elems) from global
// to LDS.  D# per ISA ch.8: group0 {count=1, lds_addr, global_addr, type=2},
// group1 {data_size, tensor_dim0/1 == tile_dim0/1 (no OOB), dim0 stride}.
// LDS flat addresses truncate to the LDS byte offset in bits [31:0] (ISA 10.2).
// 6-arg builtin form (clang-23 lane): (g0, g1, g2, g3, g_extra, cpol).
// ---------------------------------------------------------------------------
__device__ inline void tdm_load_2d(const bf16* gaddr, bf16* laddr,
                                   unsigned d0_elems, unsigned d1_rows,
                                   unsigned stride_elems) {
  unsigned long long ga = (unsigned long long)(uintptr_t)gaddr;
  unsigned la = (unsigned)(uintptr_t)laddr;
  v4u32 g0;
  g0[0] = 1u;                                        // count=1, user D#
  g0[1] = la;                                        // lds_addr (bytes)
  g0[2] = (unsigned)(ga & 0xffffffffu);              // global_addr[31:0]
  g0[3] = (unsigned)((ga >> 32) & 0x1ffffffu) | (2u << 30);  // addr[56:32] | type=2
  v8i32 g1;
  g1[0] = (int)(1u << 16);                           // wg_mask=0, data_size=1 (2B)
  g1[1] = (int)((d0_elems & 0xffffu) << 16);         // tensor_dim0[15:0] @ bit48
  g1[2] = (int)(((d0_elems >> 16) & 0xffffu) | ((d1_rows & 0xffffu) << 16));
  g1[3] = (int)(((d1_rows >> 16) & 0xffffu) | ((d0_elems & 0xffffu) << 16)); // tile_dim0
  g1[4] = (int)(d1_rows & 0xffffu);                  // tile_dim1 (tile_dim2=0)
  g1[5] = (int)stride_elems;                         // tensor_dim0_stride[31:0]
  g1[6] = 0;
  g1[7] = 0;
  v4i32 gz; gz[0] = gz[1] = gz[2] = gz[3] = 0;
  v8i32 gz8;
#pragma unroll
  for (int i = 0; i < 8; i++) gz8[i] = 0;
  __builtin_amdgcn_tensor_load_to_lds(g0, g1, gz, gz, gz8, 0);
}

// ---------------------------------------------------------------------------
// conversions
// ---------------------------------------------------------------------------
__global__ void k_cvt_x(const float* __restrict__ x, bf16* __restrict__ xb, int n) {
  int i = blockIdx.x * 256 + threadIdx.x;
  if (i < n) xb[i] = (bf16)x[i];
}

// Wfx/Wx (OC,IC,3,3) -> Wt[oc][tap*256+ic]  (row-major [N=oc][K])
__global__ void k_cvt_wconv(const float* __restrict__ Wfx, const float* __restrict__ Wx,
                            bf16* __restrict__ Wft, bf16* __restrict__ Wxt) {
  int i = blockIdx.x * 256 + threadIdx.x;
  if (i >= 512 * 2304) return;
  int oc = i / 2304, r = i % 2304;
  int tap = r >> 8, ic = r & 255;
  int src = (oc * 256 + ic) * 9 + tap;
  Wft[i] = (bf16)Wfx[src];
  Wxt[i] = (bf16)Wx[src];
}

// Wout (512,256) -> Wot[n][k] (256 x 512)
__global__ void k_cvt_wout(const float* __restrict__ Wout, bf16* __restrict__ Wot) {
  int i = blockIdx.x * 256 + threadIdx.x;
  if (i >= 256 * 512) return;
  int n = i >> 9, k = i & 511;
  Wot[i] = (bf16)Wout[k * 256 + n];
}

// ---------------------------------------------------------------------------
// 3x3 conv (SAME) as implicit GEMM, bf16 WMMA, f32 accum.
// grid.x = wtile(2) * h(128) * b(4); grid.y = head (oc tile of 64); block=128.
// Output: out[b][head][pixel][c] bf16.  (Manual staging: rows need zero-pad.)
// ---------------------------------------------------------------------------
__global__ void __launch_bounds__(128) k_conv(const bf16* __restrict__ xb,
                                              const bf16* __restrict__ Wt,
                                              const float* __restrict__ bias,
                                              bf16* __restrict__ outp) {
  __shared__ __align__(16) bf16 At[64 * 32];
  __shared__ __align__(16) bf16 Bt[64 * 32];
  int gx = blockIdx.x;
  int wt = gx & 1, h = (gx >> 1) & 127, b = gx >> 8;
  int head = blockIdx.y, oc0 = head * 64, w0 = wt * 64;
  int t = threadIdx.x, srow = t >> 1, shalf = t & 1;

  v8f acc[4]; acc_zero(acc);
  uint4 z4 = make_uint4(0u, 0u, 0u, 0u);

  for (int tap = 0; tap < 9; tap++) {
    int dy = tap / 3 - 1, dx = tap % 3 - 1;
    int sh = h + dy;
    int sw = w0 + srow + dx;
    bool inb = (sh >= 0 && sh < 128 && sw >= 0 && sw < 128);
    const bf16* ap = xb + (((size_t)(b * 128 + sh) * 128 + sw) * 256);
    const bf16* wp = Wt + (size_t)(oc0 + srow) * 2304 + tap * 256;
    for (int kc = 0; kc < 8; kc++) {
      __syncthreads();
      int ldo = srow * 32 + shalf * 16;
      if (inb) {
        const uint4* s = (const uint4*)(ap + kc * 32 + shalf * 16);
        *(uint4*)&At[ldo]     = s[0];
        *(uint4*)&At[ldo + 8] = s[1];
      } else {
        *(uint4*)&At[ldo]     = z4;
        *(uint4*)&At[ldo + 8] = z4;
      }
      const uint4* ws = (const uint4*)(wp + kc * 32 + shalf * 16);
      *(uint4*)&Bt[ldo]     = ws[0];
      *(uint4*)&Bt[ldo + 8] = ws[1];
      __syncthreads();
      wmma_tile4(At, Bt, WAVE * 16, acc);
    }
  }
  int half = (LANE >> 4), ncol = LANE & 15;
#pragma unroll
  for (int nt = 0; nt < 4; nt++)
#pragma unroll
    for (int r = 0; r < 8; r++) {
      int oc = oc0 + nt * 16 + ncol;
      float v = acc[nt][r] + bias[oc];
      int m = WAVE * 16 + r + 8 * half;
      int pix = h * 128 + w0 + m;
      outp[((size_t)(b * 8 + head) * NPIX + pix) * 64 + nt * 16 + ncol] = (bf16)v;
    }
}

// ---------------------------------------------------------------------------
// slice weights: sw = softmax((xm @ Wslice + bslice)/temp)  (VALU; tiny GEMM)
// ---------------------------------------------------------------------------
__global__ void k_slice(const bf16* __restrict__ xmbf, const float* __restrict__ Wslice,
                        const float* __restrict__ bslice, const float* __restrict__ temp,
                        bf16* __restrict__ swbf) {
  size_t gid = (size_t)blockIdx.x * 256 + threadIdx.x;  // < 4*8*16384
  int h = (int)((gid >> 14) & 7);
  const bf16* xp = xmbf + gid * 64;
  float a[64];
#pragma unroll
  for (int g = 0; g < 64; g++) a[g] = bslice[g];
  for (int d = 0; d < 64; d++) {
    float xv = (float)xp[d];
    const float* wr = Wslice + d * 64;
#pragma unroll
    for (int g = 0; g < 64; g++) a[g] += xv * wr[g];
  }
  float tt = fminf(fmaxf(temp[h], 0.1f), 5.0f);
  float ti = 1.0f / tt;
  float mx = -1e30f;
#pragma unroll
  for (int g = 0; g < 64; g++) { a[g] *= ti; mx = fmaxf(mx, a[g]); }
  float s = 0.0f;
#pragma unroll
  for (int g = 0; g < 64; g++) { a[g] = __expf(a[g] - mx); s += a[g]; }
  float inv = 1.0f / s;
  bf16* op = swbf + gid * 64;
#pragma unroll
  for (int g = 0; g < 64; g++) op[g] = (bf16)(a[g] * inv);
}

// snorm[b,h,g] = sum_n sw[b,h,n,g]
__global__ void k_snorm(const bf16* __restrict__ swbf, float* __restrict__ snorm) {
  int bhg = blockIdx.x;
  int bh = bhg >> 6, g = bhg & 63;
  const bf16* p = swbf + (size_t)bh * NPIX * 64 + g;
  float s = 0.0f;
  for (int n = threadIdx.x; n < NPIX; n += 256) s += (float)p[(size_t)n * 64];
  __shared__ float red[256];
  red[threadIdx.x] = s; __syncthreads();
  for (int o = 128; o > 0; o >>= 1) {
    if (threadIdx.x < o) red[threadIdx.x] += red[threadIdx.x + o];
    __syncthreads();
  }
  if (threadIdx.x == 0) snorm[bhg] = red[0];
}

// ---------------------------------------------------------------------------
// st[g,c] = (SW^T @ FX)/(snorm+1e-5): per (b,h) one WG, 64x64 out, K=16384
// (both operands need transpose -> manual staging; prefetch next K panel)
// ---------------------------------------------------------------------------
__global__ void __launch_bounds__(128) k_st(const bf16* __restrict__ swbf,
                                            const bf16* __restrict__ fxbf,
                                            const float* __restrict__ snorm,
                                            float* __restrict__ st) {
  __shared__ __align__(16) bf16 At[64 * 32];
  __shared__ __align__(16) bf16 Bt[64 * 32];
  int bh = blockIdx.x;
  const bf16* swp = swbf + (size_t)bh * NPIX * 64;
  const bf16* fxp = fxbf + (size_t)bh * NPIX * 64;
  int t = threadIdx.x, j = t & 31, cb = (t >> 5) * 16;
  v8f acc[4]; acc_zero(acc);

  for (int kc = 0; kc < NPIX / 32; kc++) {
    int n0 = kc * 32;
    uint4 a0 = *(const uint4*)(swp + (size_t)(n0 + j) * 64 + cb);
    uint4 a1 = *(const uint4*)(swp + (size_t)(n0 + j) * 64 + cb + 8);
    uint4 b0 = *(const uint4*)(fxp + (size_t)(n0 + j) * 64 + cb);
    uint4 b1 = *(const uint4*)(fxp + (size_t)(n0 + j) * 64 + cb + 8);
    // prefetch next K panel rows (global_prefetch_b8)
    __builtin_prefetch(swp + (size_t)(n0 + 32 + j) * 64, 0, 0);
    __builtin_prefetch(fxp + (size_t)(n0 + 32 + j) * 64, 0, 0);
    __syncthreads();
    const bf16* av = (const bf16*)&a0;
#pragma unroll
    for (int i = 0; i < 8; i++) At[(cb + i) * 32 + j] = av[i];
    av = (const bf16*)&a1;
#pragma unroll
    for (int i = 0; i < 8; i++) At[(cb + 8 + i) * 32 + j] = av[i];
    const bf16* bv = (const bf16*)&b0;
#pragma unroll
    for (int i = 0; i < 8; i++) Bt[(cb + i) * 32 + j] = bv[i];
    bv = (const bf16*)&b1;
#pragma unroll
    for (int i = 0; i < 8; i++) Bt[(cb + 8 + i) * 32 + j] = bv[i];
    __syncthreads();
    wmma_tile4(At, Bt, WAVE * 16, acc);
  }
  const float* sn = snorm + bh * 64;
  float* stp = st + (size_t)bh * 64 * 64;
  int half = (LANE >> 4), ncol = LANE & 15;
#pragma unroll
  for (int nt = 0; nt < 4; nt++)
#pragma unroll
    for (int r = 0; r < 8; r++) {
      int g = WAVE * 16 + r + 8 * half;
      float sc = 1.0f / (sn[g] + 1e-5f);
      stp[g * 64 + nt * 16 + ncol] = acc[nt][r] * sc;
    }
}

// ---------------------------------------------------------------------------
// Erwin ball-transformer: one workgroup per ball-tree (bh), all-LDS, f32 VALU
// ---------------------------------------------------------------------------
struct BlkP { const float *g1,*b1,*peW,*peB,*Wqkv,*bqkv,*Wo,*bo,*sigma,*g2,*b2,*W1,*bm1,*W2,*bm2; };
struct ErwinP {
  const float *embedW, *embedB;
  BlkP enc0[2];
  const float *poolW, *poolB;
  BlkP enc1[2];
  const float *unpoolW, *unpoolB;
  BlkP dec0[2];
};

__device__ inline float wred32(float v) {
#pragma unroll
  for (int o = 16; o > 0; o >>= 1) v += __shfl_xor(v, o);
  return v;
}
__device__ inline float geluf(float x) {
  return 0.5f * x * (1.0f + tanhf(0.7978845608028654f * (x + 0.044715f * x * x * x)));
}

// stable odd-even argsort of n<=64 keys in LDS
__device__ void argsortLDS(const float* pos, int coord, int n, int* idx) {
  int tid = threadIdx.x;
  for (int i = tid; i < n; i += 256) idx[i] = i;
  __syncthreads();
  for (int r = 0; r < n; r++) {
    int i = 2 * tid + (r & 1);
    if (tid < n / 2 && i + 1 < n) {
      int a = idx[i], b = idx[i + 1];
      if (pos[a * 2 + coord] > pos[b * 2 + coord]) { idx[i] = b; idx[i + 1] = a; }
    }
    __syncthreads();
  }
}

template <int C, int BALL, int NH>
__device__ void ball_block(const BlkP p, float* Xl, const float* posl,
                           float* Hl, float* Obuf, float* BMs) {
  constexpr int G = 2 * BALL, HD = C / NH, CPL = C / 32, WDT = 4 * C;
  int tid = threadIdx.x, wv = tid >> 5, ln = tid & 31;

  // ball centers
  if (tid < 4) {
    int bi = tid >> 1, k = tid & 1;
    float s = 0.0f;
    for (int i = 0; i < BALL; i++) s += posl[(bi * BALL + i) * 2 + k];
    BMs[tid] = s / (float)BALL;
  }
  __syncthreads();

  // h = LN(x)*g1+b1 + rel@peW + peB
  for (int g = wv; g < G; g += 8) {
    float vals[CPL]; float s = 0.0f;
#pragma unroll
    for (int i = 0; i < CPL; i++) { vals[i] = Xl[g * C + i * 32 + ln]; s += vals[i]; }
    s = wred32(s); float mean = s / (float)C;
    float v2 = 0.0f;
#pragma unroll
    for (int i = 0; i < CPL; i++) { float d = vals[i] - mean; v2 += d * d; }
    v2 = wred32(v2) / (float)C;
    float rs = rsqrtf(v2 + 1e-5f);
    int bi = g / BALL;
    float r0 = posl[g * 2] - BMs[bi * 2], r1 = posl[g * 2 + 1] - BMs[bi * 2 + 1];
#pragma unroll
    for (int i = 0; i < CPL; i++) {
      int j = i * 32 + ln;
      Hl[g * C + j] = (vals[i] - mean) * rs * p.g1[j] + p.b1[j]
                      + r0 * p.peW[j] + r1 * p.peW[C + j] + p.peB[j];
    }
  }
  __syncthreads();

  // attention within balls (wave per (ball,head) combo, lane = token)
  constexpr int NCB = 2 * NH;
  for (int cb = wv; cb < NCB; cb += 8) {
    int bl = cb / NH, hh = cb % NH, t0 = bl * BALL;
    bool act = (ln < BALL);
    int t = t0 + (act ? ln : 0);
    float q[HD], k[HD], v[HD];
#pragma unroll
    for (int d = 0; d < HD; d++) {
      int qc = hh * HD + d;
      float aq = p.bqkv[qc], ak = p.bqkv[C + qc], av = p.bqkv[2 * C + qc];
      for (int j = 0; j < C; j++) {
        float hv = Hl[t * C + j];
        const float* wr = p.Wqkv + j * 3 * C;
        aq += hv * wr[qc]; ak += hv * wr[C + qc]; av += hv * wr[2 * C + qc];
      }
      q[d] = aq; k[d] = ak; v[d] = av;
    }
    float sig = p.sigma[hh];
    float px = posl[t * 2], py = posl[t * 2 + 1];
    const float scl = rsqrtf((float)HD);
    float lg[BALL]; float mx = -1e30f;
#pragma unroll
    for (int j = 0; j < BALL; j++) {
      float a = 0.0f;
#pragma unroll
      for (int d = 0; d < HD; d++) a += q[d] * __shfl(k[d], j);
      float dx = px - posl[(t0 + j) * 2], dyv = py - posl[(t0 + j) * 2 + 1];
      a = a * scl + sig * sqrtf(dx * dx + dyv * dyv);
      lg[j] = a; mx = fmaxf(mx, a);
    }
    float se = 0.0f;
#pragma unroll
    for (int j = 0; j < BALL; j++) { lg[j] = __expf(lg[j] - mx); se += lg[j]; }
    float inv = 1.0f / se;
    float o[HD];
#pragma unroll
    for (int d = 0; d < HD; d++) o[d] = 0.0f;
#pragma unroll
    for (int j = 0; j < BALL; j++) {
      float a = lg[j] * inv;
#pragma unroll
      for (int d = 0; d < HD; d++) o[d] += a * __shfl(v[d], j);
    }
    if (act) {
#pragma unroll
      for (int d = 0; d < HD; d++) Obuf[t * C + hh * HD + d] = o[d];
    }
  }
  __syncthreads();

  // x += o @ Wo + bo
  for (int e = tid; e < G * C; e += 256) {
    int g = e / C, j = e % C;
    float a = p.bo[j];
    for (int kk = 0; kk < C; kk++) a += Obuf[g * C + kk] * p.Wo[kk * C + j];
    Xl[e] += a;
  }
  __syncthreads();

  // x += gelu(LN(x)@W1+bm1) @ W2 + bm2   (wave per row, hidden in registers)
  for (int g = wv; g < G; g += 8) {
    float vals[CPL]; float s = 0.0f;
#pragma unroll
    for (int i = 0; i < CPL; i++) { vals[i] = Xl[g * C + i * 32 + ln]; s += vals[i]; }
    s = wred32(s); float mean = s / (float)C;
    float v2 = 0.0f;
#pragma unroll
    for (int i = 0; i < CPL; i++) { float d = vals[i] - mean; v2 += d * d; }
    v2 = wred32(v2) / (float)C;
    float rs = rsqrtf(v2 + 1e-5f);
#pragma unroll
    for (int i = 0; i < CPL; i++) {
      int j = i * 32 + ln;
      Hl[g * C + j] = (vals[i] - mean) * rs * p.g2[j] + p.b2[j];
    }
    float hid[WDT / 32];
#pragma unroll
    for (int kk = 0; kk < WDT / 32; kk++) {
      int k2 = kk * 32 + ln;
      float a = p.bm1[k2];
      for (int j = 0; j < C; j++) a += Hl[g * C + j] * p.W1[j * WDT + k2];
      hid[kk] = geluf(a);
    }
    float accv[CPL];
#pragma unroll
    for (int i = 0; i < CPL; i++) accv[i] = p.bm2[i * 32 + ln];
    for (int kk = 0; kk < WDT / 32; kk++) {
#pragma unroll
      for (int sl = 0; sl < 32; sl++) {
        float hv = __shfl(hid[kk], sl);
        const float* wr = p.W2 + (kk * 32 + sl) * C;
#pragma unroll
        for (int i = 0; i < CPL; i++) accv[i] += hv * wr[i * 32 + ln];
      }
    }
#pragma unroll
    for (int i = 0; i < CPL; i++) Xl[g * C + i * 32 + ln] += accv[i];
  }
  __syncthreads();
}

template <int C, int BALL, int NH>
__device__ void layer2(const BlkP* blks, float* Xl, float* posl, float* Hl, float* Obuf,
                       int* prm, float* posR, float* BMs) {
  constexpr int G = 2 * BALL;
  int tid = threadIdx.x;
  ball_block<C, BALL, NH>(blks[0], Xl, posl, Hl, Obuf, BMs);
  argsortLDS(posl, 1, G, prm);
  float tmp[16];
#pragma unroll
  for (int i = 0; i < 16; i++) {
    int e = tid + 256 * i; int g = e / C, j = e % C;
    tmp[i] = Xl[prm[g] * C + j];
  }
  if (tid < 2 * G) posR[tid] = posl[prm[tid >> 1] * 2 + (tid & 1)];
  __syncthreads();
#pragma unroll
  for (int i = 0; i < 16; i++) Xl[tid + 256 * i] = tmp[i];
  __syncthreads();
  ball_block<C, BALL, NH>(blks[1], Xl, posR, Hl, Obuf, BMs);
#pragma unroll
  for (int i = 0; i < 16; i++) tmp[i] = Xl[tid + 256 * i];
  __syncthreads();
#pragma unroll
  for (int i = 0; i < 16; i++) {
    int e = tid + 256 * i; int g = e / C, j = e % C;
    Xl[prm[g] * C + j] = tmp[i];
  }
  __syncthreads();
}

__global__ void __launch_bounds__(256) k_erwin(const float* __restrict__ st,
                                               const float* __restrict__ posG,
                                               ErwinP P, bf16* __restrict__ otbf) {
  __shared__ float X[4096];
  __shared__ float Hb[4096];
  __shared__ float X1[4096];
  __shared__ float POSR[128];
  __shared__ float PS[128];
  __shared__ float PRb[128];
  __shared__ float P1[64];
  __shared__ float PP[64];
  __shared__ int PERM0[64];
  __shared__ int PRMR[64];
  __shared__ int PERM1[32];
  __shared__ float BMs[4];

  int bh = blockIdx.x, tid = threadIdx.x;
  const float* tok = st + (size_t)bh * 4096;
  const float* pos = posG + (size_t)bh * 128;

  // linear embed into Hb (unsorted)
  for (int e = tid; e < 4096; e += 256) {
    int g = e >> 6, j = e & 63;
    float a = P.embedB[j];
    for (int k2 = 0; k2 < 64; k2++) a += tok[g * 64 + k2] * P.embedW[k2 * 64 + j];
    Hb[e] = a;
  }
  if (tid < 128) POSR[tid] = pos[tid];
  __syncthreads();
  argsortLDS(POSR, 0, 64, PERM0);
  for (int e = tid; e < 4096; e += 256) {
    int g = e >> 6, j = e & 63;
    X[e] = Hb[PERM0[g] * 64 + j];
  }
  if (tid < 128) PS[tid] = POSR[PERM0[tid >> 1] * 2 + (tid & 1)];
  __syncthreads();

  layer2<64, 32, 4>(P.enc0, X, PS, Hb, X1, PRMR, PRb, BMs);

  float sk[16];
#pragma unroll
  for (int i = 0; i < 16; i++) sk[i] = X[tid + 256 * i];

  // pool: (64,64)->(32,128) @ poolW; pooled positions
  for (int e = tid; e < 4096; e += 256) {
    int g = e >> 7, j = e & 127;
    float a = P.poolB[j];
    for (int k2 = 0; k2 < 128; k2++) a += X[g * 128 + k2] * P.poolW[k2 * 128 + j];
    Hb[e] = a;
  }
  if (tid < 64) PP[tid] = 0.5f * (PS[(tid >> 1) * 4 + (tid & 1)] + PS[(tid >> 1) * 4 + 2 + (tid & 1)]);
  __syncthreads();
  argsortLDS(PP, 0, 32, PERM1);
  for (int e = tid; e < 4096; e += 256) {
    int g = e >> 7, j = e & 127;
    X1[e] = Hb[PERM1[g] * 128 + j];
  }
  if (tid < 64) P1[tid] = PP[PERM1[tid >> 1] * 2 + (tid & 1)];
  __syncthreads();

  layer2<128, 16, 8>(P.enc1, X1, P1, Hb, X, PRMR, PRb, BMs);

  // unsort pooled, unpool + skip
  for (int e = tid; e < 4096; e += 256) {
    int g = e >> 7, j = e & 127;
    Hb[PERM1[g] * 128 + j] = X1[e];
  }
  __syncthreads();
#pragma unroll
  for (int i = 0; i < 16; i++) {
    int e = tid + 256 * i; int gp = e >> 7, jp = e & 127;
    float a = P.unpoolB[jp];
    for (int k2 = 0; k2 < 128; k2++) a += Hb[gp * 128 + k2] * P.unpoolW[k2 * 128 + jp];
    X[e] = sk[i] + a;
  }
  __syncthreads();

  layer2<64, 32, 4>(P.dec0, X, PS, Hb, X1, PRMR, PRb, BMs);

  // undo the initial sort, emit bf16 tokens
  for (int e = tid; e < 4096; e += 256) {
    int g = e >> 6, j = e & 63;
    otbf[(size_t)bh * 4096 + PERM0[g] * 64 + j] = (bf16)X[e];
  }
}

// ---------------------------------------------------------------------------
// deslice: ox[b,n,h*64+c] = SW[n,:] @ OT[:,c]  (per (b,h): M=16384,N=64,K=64)
// A tile (contiguous sw rows) via TDM; B tile needs transpose -> manual.
// ---------------------------------------------------------------------------
__global__ void __launch_bounds__(128) k_deslice(const bf16* __restrict__ swbf,
                                                 const bf16* __restrict__ otbf,
                                                 bf16* __restrict__ oxbf) {
  __shared__ __align__(16) bf16 At[64 * 32];
  __shared__ __align__(16) bf16 Bt[64 * 32];
  int mt = blockIdx.x;          // 256 tiles of 64 pixels
  int bh = blockIdx.y; int b = bh >> 3, h = bh & 7;
  const bf16* swp = swbf + (size_t)bh * NPIX * 64;
  const bf16* otp = otbf + (size_t)bh * 64 * 64;
  int t = threadIdx.x, m0 = mt * 64;
  v8f acc[4]; acc_zero(acc);

  for (int kc = 0; kc < 2; kc++) {
    __syncthreads();
    if (t < 32) {  // one wave issues the TDM copy: A tile = 64 rows x 32 bf16
      tdm_load_2d(swp + (size_t)m0 * 64 + kc * 32, At, 32u, 64u, 64u);
      __builtin_amdgcn_s_wait_tensorcnt(0);
    }
    {
      int j = t & 31, cb = (t >> 5) * 16;
      uint4 b0 = *(const uint4*)(otp + (size_t)(kc * 32 + j) * 64 + cb);
      uint4 b1 = *(const uint4*)(otp + (size_t)(kc * 32 + j) * 64 + cb + 8);
      const bf16* bv = (const bf16*)&b0;
#pragma unroll
      for (int i = 0; i < 8; i++) Bt[(cb + i) * 32 + j] = bv[i];
      bv = (const bf16*)&b1;
#pragma unroll
      for (int i = 0; i < 8; i++) Bt[(cb + 8 + i) * 32 + j] = bv[i];
    }
    __syncthreads();
    wmma_tile4(At, Bt, WAVE * 16, acc);
  }
  int half = (LANE >> 4), ncol = LANE & 15;
#pragma unroll
  for (int nt = 0; nt < 4; nt++)
#pragma unroll
    for (int r = 0; r < 8; r++) {
      int m = WAVE * 16 + r + 8 * half;
      int n = m0 + m, c = nt * 16 + ncol;
      oxbf[((size_t)b * NPIX + n) * INNER + h * 64 + c] = (bf16)acc[nt][r];
    }
}

// ---------------------------------------------------------------------------
// out projection: y = ox(65536x512) @ Wout(512x256) + bout, f32 output.
// Both tiles are straight 2D copies -> Tensor Data Mover for A and B.
// ---------------------------------------------------------------------------
__global__ void __launch_bounds__(128) k_out(const bf16* __restrict__ oxbf,
                                             const bf16* __restrict__ wot,
                                             const float* __restrict__ bout,
                                             float* __restrict__ y) {
  __shared__ __align__(16) bf16 At[64 * 32];
  __shared__ __align__(16) bf16 Bt[64 * 32];
  int m0 = blockIdx.x * 64, n0 = blockIdx.y * 64;
  int t = threadIdx.x;
  v8f acc[4]; acc_zero(acc);

  for (int kc = 0; kc < 16; kc++) {
    __syncthreads();
    if (t < 32) {  // TDM: A tile and B tile, each 64 rows x 32 bf16, stride 512
      tdm_load_2d(oxbf + (size_t)m0 * 512 + kc * 32, At, 32u, 64u, 512u);
      tdm_load_2d(wot + (size_t)n0 * 512 + kc * 32, Bt, 32u, 64u, 512u);
      __builtin_amdgcn_s_wait_tensorcnt(0);
    }
    __syncthreads();
    wmma_tile4(At, Bt, WAVE * 16, acc);
  }
  int half2 = (LANE >> 4), ncol = LANE & 15;
#pragma unroll
  for (int nt = 0; nt < 4; nt++)
#pragma unroll
    for (int r = 0; r < 8; r++) {
      int m = WAVE * 16 + r + 8 * half2;
      int n = n0 + nt * 16 + ncol;
      y[(size_t)(m0 + m) * 256 + n] = acc[nt][r] + bout[n];
    }
}

// ---------------------------------------------------------------------------
// host side
// ---------------------------------------------------------------------------
static void fill_blk(BlkP& b, void* const* d, int base) {
  const float** f = (const float**)&b;
  for (int i = 0; i < 15; i++) f[i] = (const float*)d[base + i];
}

extern "C" void kernel_launch(void* const* d_in, const int* in_sizes, int n_in,
                              void* d_out, int out_size, void* d_ws, size_t ws_size,
                              hipStream_t stream) {
  (void)in_sizes; (void)n_in; (void)out_size; (void)ws_size;
  const float* x      = (const float*)d_in[0];
  const float* pos    = (const float*)d_in[1];
  const float* Wfx    = (const float*)d_in[2];
  const float* bfx    = (const float*)d_in[3];
  const float* Wx     = (const float*)d_in[4];
  const float* bx     = (const float*)d_in[5];
  const float* Wslice = (const float*)d_in[6];
  const float* bslice = (const float*)d_in[7];
  const float* temp   = (const float*)d_in[8];
  const float* Wout   = (const float*)d_in[9];
  const float* bout   = (const float*)d_in[10];

  ErwinP P;
  P.embedW = (const float*)d_in[11];
  P.embedB = (const float*)d_in[12];
  fill_blk(P.enc0[0], d_in, 13);
  fill_blk(P.enc0[1], d_in, 28);
  P.poolW = (const float*)d_in[43];
  P.poolB = (const float*)d_in[44];
  fill_blk(P.enc1[0], d_in, 45);
  fill_blk(P.enc1[1], d_in, 60);
  P.unpoolW = (const float*)d_in[75];
  P.unpoolB = (const float*)d_in[76];
  fill_blk(P.dec0[0], d_in, 77);
  fill_blk(P.dec0[1], d_in, 92);

  char* ws = (char*)d_ws; size_t off = 0;
  auto take = [&](size_t bytes) -> char* {
    char* r = ws + off;
    off = (off + bytes + 255) & ~(size_t)255;
    return r;
  };
  bf16* xbf   = (bf16*)take((size_t)BN * NPIX * DIMi * 2);      // 33.5 MB
  bf16* wft   = (bf16*)take((size_t)512 * 2304 * 2);
  bf16* wxt   = (bf16*)take((size_t)512 * 2304 * 2);
  bf16* wot   = (bf16*)take((size_t)256 * 512 * 2);
  bf16* fxbf  = (bf16*)take((size_t)BN * NPIX * INNER * 2);     // 67 MB
  bf16* xmbf  = (bf16*)take((size_t)BN * NPIX * INNER * 2);     // 67 MB (reused as ox)
  bf16* oxbf  = xmbf;                                           // alias: xm dead after k_slice
  bf16* swbf  = (bf16*)take((size_t)BN * HEADS * NPIX * GS * 2);// 67 MB
  float* snorm = (float*)take((size_t)BN * HEADS * GS * 4);
  float* stb   = (float*)take((size_t)BN * HEADS * GS * DH * 4);
  bf16* otbf   = (bf16*)take((size_t)BN * HEADS * GS * DH * 2);

  int nx = BN * NPIX * DIMi;
  k_cvt_x<<<(nx + 255) / 256, 256, 0, stream>>>(x, xbf, nx);
  k_cvt_wconv<<<(512 * 2304 + 255) / 256, 256, 0, stream>>>(Wfx, Wx, wft, wxt);
  k_cvt_wout<<<(256 * 512 + 255) / 256, 256, 0, stream>>>(Wout, wot);

  k_conv<<<dim3(1024, 8, 1), 128, 0, stream>>>(xbf, wft, bfx, fxbf);
  k_conv<<<dim3(1024, 8, 1), 128, 0, stream>>>(xbf, wxt, bx, xmbf);

  k_slice<<<(BN * HEADS * NPIX) / 256, 256, 0, stream>>>(xmbf, Wslice, bslice, temp, swbf);
  k_snorm<<<BN * HEADS * GS, 256, 0, stream>>>(swbf, snorm);
  k_st<<<BN * HEADS, 128, 0, stream>>>(swbf, fxbf, snorm, stb);

  k_erwin<<<BN * HEADS, 256, 0, stream>>>(stb, pos, P, otbf);

  k_deslice<<<dim3(NPIX / 64, BN * HEADS, 1), 128, 0, stream>>>(swbf, otbf, oxbf);
  k_out<<<dim3(BN * NPIX / 64, 4, 1), 128, 0, stream>>>(oxbf, wot, bout, (float*)d_out);
}
